// GCNWithAttention_40020505264514
// MI455X (gfx1250) — matrix operand order repaired and checked by reading, not compiled, not run on hardware
//
#include <hip/hip_runtime.h>

typedef __attribute__((ext_vector_type(16))) _Float16 v16h;
typedef __attribute__((ext_vector_type(8)))  _Float16 v8h;
typedef __attribute__((ext_vector_type(8)))  float    v8f;

#define XH_LD   512   // f16 activation buffer row stride (halves)
#define XCAT_LD 464   // concat buffer row stride (floats), 456 used
#define WT_LD   512   // transposed-weight K stride (halves)
#define T_LD    400   // t buffer row stride (floats)
#define TH_LD   512   // packed f16 t buffer row stride (halves)
#define H_LD    256   // hidden width

// load 16 halves: p[0..7] and (p+step)[0..7] -> one v16h fragment
__device__ __forceinline__ v16h ld_frag(const _Float16* p, int step) {
  v8h l0 = *(const v8h*)p;
  v8h l1 = *(const v8h*)(p + step);
  return __builtin_shufflevector(l0, l1, 0,1,2,3,4,5,6,7,8,9,10,11,12,13,14,15);
}

// ---------------------------------------------------------------------------
// Fast WMMA GEMM: C[M,Nstore] = act((A @ B^T_stored + bias) * scale)
// Requirements (enforced by caller via padding):
//   M % 16 == 0, Kd % 32 == 0
//   A: M x >=Kd row-major f16 (lda), valid (zero-padded) for all k < Kd
//   BT: B stored transposed as Npad x >=Kd f16 (ldb), Npad >= 64*strips_n,
//       rows >= Nstore zero-padded
// One wave computes a 16x64 strip: 1 A fragment x 4 B fragments, manually
// software-pipelined; only the store is column-guarded.
// A/B use contiguous k = 16*(lane/16) + s here (same order for A and B).
// ---------------------------------------------------------------------------
__global__ void k_gemm_fast(const _Float16* __restrict__ A, int lda,
                            const _Float16* __restrict__ BT, int ldb,
                            float* __restrict__ C, int ldc,
                            const float* __restrict__ bias,
                            const float* __restrict__ scale_ptr,
                            int M, int Nstore, int strips_n, int Kd,
                            int relu)
{
  const int lane = threadIdx.x & 31;
  const int wid  = blockIdx.x * (blockDim.x >> 5) + (threadIdx.x >> 5);
  const int tiles_m = M >> 4;
  if (wid >= tiles_m * strips_n) return;          // uniform per wave
  const int m0 = (wid / strips_n) << 4;
  const int n0 = (wid % strips_n) << 6;

  const int l15   = lane & 15;
  const int hi    = (lane & 16) ? 1 : 0;
  const int koffA = hi ? 16 : 0;                  // A: K halves 0..15 / 16..31
  const int koffB = hi ? 16 : 0;                  // B: K halves 0..15 / 16..31

  const _Float16* ap  = A  + (size_t)(m0 + l15) * lda + koffA;
  const _Float16* bp0 = BT + (size_t)(n0 + l15) * ldb + koffB;
  const _Float16* bp1 = bp0 + (size_t)16 * ldb;
  const _Float16* bp2 = bp0 + (size_t)32 * ldb;
  const _Float16* bp3 = bp0 + (size_t)48 * ldb;

  v8f acc0 = {}, acc1 = {}, acc2 = {}, acc3 = {};

  v16h a  = ld_frag(ap,  8);
  v16h b0 = ld_frag(bp0, 8);
  v16h b1 = ld_frag(bp1, 8);
  v16h b2 = ld_frag(bp2, 8);
  v16h b3 = ld_frag(bp3, 8);

  for (int k0 = 0; k0 < Kd; k0 += 32) {
    v16h an = a, b0n = b0, b1n = b1, b2n = b2, b3n = b3;
    const int kn = k0 + 32;
    if (kn < Kd) {                                // uniform branch
      an  = ld_frag(ap  + kn, 8);
      b0n = ld_frag(bp0 + kn, 8);
      b1n = ld_frag(bp1 + kn, 8);
      b2n = ld_frag(bp2 + kn, 8);
      b3n = ld_frag(bp3 + kn, 8);
    }
    acc0 = __builtin_amdgcn_wmma_f32_16x16x32_f16(false, a, false, b0, (short)0, acc0, false, false);
    acc1 = __builtin_amdgcn_wmma_f32_16x16x32_f16(false, a, false, b1, (short)0, acc1, false, false);
    acc2 = __builtin_amdgcn_wmma_f32_16x16x32_f16(false, a, false, b2, (short)0, acc2, false, false);
    acc3 = __builtin_amdgcn_wmma_f32_16x16x32_f16(false, a, false, b3, (short)0, acc3, false, false);
    a = an; b0 = b0n; b1 = b1n; b2 = b2n; b3 = b3n;
  }

  const float sc   = scale_ptr ? scale_ptr[0] : 1.0f;
  const int  rbase = m0 + (hi ? 8 : 0);

#define STORE_TILE(ACC, TI)                                                  \
  {                                                                          \
    const int cc = n0 + ((TI) << 4) + l15;                                   \
    if (cc < Nstore) {                                                       \
      const float bv = bias ? bias[cc] : 0.0f;                               \
      _Pragma("unroll")                                                      \
      for (int r = 0; r < 8; ++r) {                                          \
        float v = ((ACC)[r] + bv) * sc;                                      \
        if (relu) v = fmaxf(v, 0.0f);                                        \
        C[(size_t)(rbase + r) * ldc + cc] = v;                               \
      }                                                                      \
    }                                                                        \
  }
  STORE_TILE(acc0, 0)
  STORE_TILE(acc1, 1)
  STORE_TILE(acc2, 2)
  STORE_TILE(acc3, 3)
#undef STORE_TILE
}

// ---------------------------------------------------------------------------
// VtZ = V^T @ Z over the node dimension (split-K, f32 atomic accumulate).
// Operates on the padded f16 t-buffer TH (TH_LD stride): V in cols 128..255,
// Z in cols 256..383, sections + 16 tail rows zero-padded -> no guards.
// M = N = 112 (7x7 tiles), one wave per 16x16 tile, grid.y = K chunks.
// ---------------------------------------------------------------------------
__global__ void k_gemm_vtz(const _Float16* __restrict__ TH_, float* __restrict__ C,
                           int kchunk, int Kd)
{
  const int lane = threadIdx.x & 31;
  const int wid  = blockIdx.x * (blockDim.x >> 5) + (threadIdx.x >> 5);
  if (wid >= 49) return;
  const int m0 = (wid / 7) << 4;
  const int n0 = (wid % 7) << 4;
  const int l15   = lane & 15;
  const int hi    = (lane & 16) ? 1 : 0;
  const int koffA = hi ? 8 : 0;
  const int koffB = hi ? 16 : 0;
  const int row = m0 + l15;
  const int col = n0 + l15;

  const int kstart = blockIdx.y * kchunk;
  int kend = kstart + kchunk; if (kend > Kd) kend = Kd;

  v8f acc = {};
  for (int k0 = kstart; k0 < kend; k0 += 32) {
    v16h a = {}, b = {};
#pragma unroll
    for (int h = 0; h < 16; ++h) {
      int ka = k0 + ((h >> 3) << 4) + koffA + (h & 7);
      a[h] = TH_[(size_t)ka * TH_LD + 128 + row];   // V^T
    }
#pragma unroll
    for (int h = 0; h < 16; ++h) {
      int kb = k0 + ((h >> 3) << 4) + koffA + (h & 7);
      b[h] = TH_[(size_t)kb * TH_LD + 256 + col];   // Z
    }
    acc = __builtin_amdgcn_wmma_f32_16x16x32_f16(false, a, false, b, (short)0, acc, false, false);
  }
  const int rbase = m0 + (hi ? 8 : 0);
#pragma unroll
  for (int r = 0; r < 8; ++r)
    atomicAdd(&C[(size_t)(rbase + r) * 112 + col], acc[r]);
}

// ----------------------------- utility kernels -----------------------------
__global__ void k_fill_f32(float* __restrict__ p, float v, long n) {
  long i = (long)blockIdx.x * blockDim.x + threadIdx.x;
  long st = (long)gridDim.x * blockDim.x;
  for (; i < n; i += st) p[i] = v;
}

__global__ void k_deg_edges(const int* __restrict__ dst, float* __restrict__ deg, int E) {
  int i = blockIdx.x * blockDim.x + threadIdx.x;
  int st = gridDim.x * blockDim.x;
  for (; i < E; i += st) atomicAdd(&deg[dst[i]], 1.0f);
}

__global__ void k_rsqrt_ip(float* __restrict__ p, int n) {
  int i = blockIdx.x * blockDim.x + threadIdx.x;
  int st = gridDim.x * blockDim.x;
  for (; i < n; i += st) p[i] = rsqrtf(fmaxf(p[i], 1.0f));
}

// f16 convert with zero padding: dst cols [0,scols) = src, [scols,pcols) = 0
__global__ void k_f32_to_f16_pad(const float* __restrict__ s, int lds, int scols,
                                 _Float16* __restrict__ d, int ldd, int pcols, long rows) {
  long total = rows * (long)pcols;
  long i = (long)blockIdx.x * blockDim.x + threadIdx.x;
  long st = (long)gridDim.x * blockDim.x;
  for (; i < total; i += st) {
    long r = i / pcols; int c = (int)(i - r * pcols);
    d[r * ldd + c] = (c < scols) ? (_Float16)s[r * lds + c] : (_Float16)0.0f;
  }
}

// pack t (N x 400 f32) into padded f16 sections: U->cols 0..127, V->128..255,
// Z->256..383 (each 100 valid + 28 zeros)
__global__ void k_pack_t(const float* __restrict__ t, _Float16* __restrict__ th, long rows) {
  long total = rows * 384;
  long i = (long)blockIdx.x * blockDim.x + threadIdx.x;
  long st = (long)gridDim.x * blockDim.x;
  for (; i < total; i += st) {
    long r = i / 384; int j = (int)(i - r * 384);
    int sec = j >> 7, cc = j & 127;
    th[r * TH_LD + j] = (cc < 100) ? (_Float16)t[r * T_LD + sec * 100 + cc]
                                   : (_Float16)0.0f;
  }
}

// weight W[K,N] -> WT[Npad, ldk] f16 transposed, zero-padded rows and cols
__global__ void k_w_transpose_f16(const float* __restrict__ w, int K, int N, int Npad,
                                  _Float16* __restrict__ wt, int ldk) {
  long total = (long)Npad * ldk;
  long i = (long)blockIdx.x * blockDim.x + threadIdx.x;
  long st = (long)gridDim.x * blockDim.x;
  for (; i < total; i += st) {
    int n = (int)(i / ldk); int k = (int)(i - (long)n * ldk);
    wt[i] = (n < N && k < K) ? (_Float16)w[(long)k * N + n] : (_Float16)0.0f;
  }
}

// edge scatter: agg[dst] += h[src] * dis[src]*dis[dst], 4 features/thread
__global__ void k_scatter(const float* __restrict__ h, const float* __restrict__ dis,
                          const int* __restrict__ src, const int* __restrict__ dst,
                          float* __restrict__ agg, int E) {
  long total = (long)E * 64;
  long i = (long)blockIdx.x * blockDim.x + threadIdx.x;
  long st = (long)gridDim.x * blockDim.x;
  for (; i < total; i += st) {
    int e  = (int)(i >> 6);
    int c4 = ((int)i & 63) << 2;
    int s = src[e], d = dst[e];
    float cf = dis[s] * dis[d];
    const float4 hv = *(const float4*)(h + (size_t)s * H_LD + c4);
    float* ap = agg + (size_t)d * H_LD + c4;
    atomicAdd(ap + 0, hv.x * cf);
    atomicAdd(ap + 1, hv.y * cf);
    atomicAdd(ap + 2, hv.z * cf);
    atomicAdd(ap + 3, hv.w * cf);
  }
}

// x_local = relu(agg + h*dis^2 (self loop) + cb) -> xcat[:, 200:456]
__global__ void k_local_fin(const float* __restrict__ agg, const float* __restrict__ h,
                            const float* __restrict__ dis, const float* __restrict__ cb,
                            float* __restrict__ xcat, int n) {
  long total = (long)n * H_LD;
  long i = (long)blockIdx.x * blockDim.x + threadIdx.x;
  long st = (long)gridDim.x * blockDim.x;
  for (; i < total; i += st) {
    long r = i >> 8; int c = (int)i & 255;
    float d2 = dis[r] * dis[r];
    float v = agg[i] + h[i] * d2 + cb[c];
    xcat[r * XCAT_LD + 200 + c] = fmaxf(v, 0.0f);
  }
}

// xcat[:, 100:200] = t[:, 300:400]
__global__ void k_copy_T(const float* __restrict__ t, float* __restrict__ xcat, int n) {
  long total = (long)n * 100;
  long i = (long)blockIdx.x * blockDim.x + threadIdx.x;
  long st = (long)gridDim.x * blockDim.x;
  for (; i < total; i += st) {
    long r = i / 100; int c = (int)(i - r * 100);
    xcat[r * XCAT_LD + 100 + c] = t[r * T_LD + 300 + c];
  }
}

// column sums of t cols 0..199 (U then V), one block per column
__global__ void k_colsum(const float* __restrict__ t, float* __restrict__ cs, int n) {
  __shared__ float sm[256];
  int c = blockIdx.x;
  float s = 0.0f;
  for (int i = threadIdx.x; i < n; i += blockDim.x) s += t[(size_t)i * T_LD + c];
  sm[threadIdx.x] = s; __syncthreads();
  for (int stp = 128; stp > 0; stp >>= 1) {
    if (threadIdx.x < stp) sm[threadIdx.x] += sm[threadIdx.x + stp];
    __syncthreads();
  }
  if (threadIdx.x == 0) cs[c] = sm[0];
}

// inv_nf = 1 / (dot(colsumU, colsumV)/n + 1e-6)
__global__ void k_nf(const float* __restrict__ cs, float* __restrict__ nfinv, int n) {
  __shared__ float sm[128];
  float s = 0.0f;
  for (int k = threadIdx.x; k < 100; k += blockDim.x) s += cs[k] * cs[100 + k];
  sm[threadIdx.x] = s; __syncthreads();
  for (int stp = 64; stp > 0; stp >>= 1) {
    if (threadIdx.x < stp) sm[threadIdx.x] += sm[threadIdx.x + stp];
    __syncthreads();
  }
  if (threadIdx.x == 0) nfinv[0] = 1.0f / (sm[0] / (float)n + 1e-6f);
}

// vtzh[n,k] = vtzf[k,n], f16, padded to 128x128 (B^T layout for res GEMM)
__global__ void k_conv_vtz(const float* __restrict__ vf, _Float16* __restrict__ vh) {
  int total = 128 * 128;
  int i = blockIdx.x * blockDim.x + threadIdx.x;
  int st = gridDim.x * blockDim.x;
  for (; i < total; i += st) {
    int n = i >> 7, k = i & 127;
    vh[i] = (n < 100 && k < 100) ? (_Float16)vf[k * 112 + n] : (_Float16)0.0f;
  }
}

// BN stats: one block per channel -> scale/shift
__global__ void k_bn_stats(const float* __restrict__ x, int n, int C_,
                           const float* __restrict__ g, const float* __restrict__ bt,
                           float* __restrict__ scale, float* __restrict__ shift) {
  __shared__ float s1[256];
  __shared__ float s2[256];
  int c = blockIdx.x;
  float a = 0.0f, b = 0.0f;
  for (int i = threadIdx.x; i < n; i += blockDim.x) {
    float v = x[(size_t)i * C_ + c];
    a += v; b += v * v;
  }
  s1[threadIdx.x] = a; s2[threadIdx.x] = b; __syncthreads();
  for (int stp = 128; stp > 0; stp >>= 1) {
    if (threadIdx.x < stp) { s1[threadIdx.x] += s1[threadIdx.x + stp];
                             s2[threadIdx.x] += s2[threadIdx.x + stp]; }
    __syncthreads();
  }
  if (threadIdx.x == 0) {
    float mean = s1[0] / (float)n;
    float var  = s2[0] / (float)n - mean * mean;
    float sc   = g[c] * rsqrtf(var + 1e-5f);
    scale[c] = sc;
    shift[c] = bt[c] - mean * sc;
  }
}

__global__ void k_bn_apply(const float* __restrict__ xb, const float* __restrict__ scale,
                           const float* __restrict__ shift, float* __restrict__ xa,
                           long total, int C_) {
  long i = (long)blockIdx.x * blockDim.x + threadIdx.x;
  long st = (long)gridDim.x * blockDim.x;
  for (; i < total; i += st) {
    int c = (int)(i % C_);
    xa[i] = xb[i] * scale[c] + shift[c];
  }
}

// ---------------------------------------------------------------------------
extern "C" void kernel_launch(void* const* d_in, const int* in_sizes, int n_in,
                              void* d_out, int out_size, void* d_ws, size_t ws_size,
                              hipStream_t stream) {
  (void)n_in; (void)out_size; (void)ws_size;
  const int N = in_sizes[0] / 128;   // 50000, multiple of 16
  const int E = in_sizes[1] / 2;
  const float* x0  = (const float*)d_in[0];
  const int* esrc  = (const int*)d_in[1];
  const int* edst  = esrc + E;

  const float* cw[3] = {(const float*)d_in[2],  (const float*)d_in[8],  (const float*)d_in[14]};
  const float* cb[3] = {(const float*)d_in[3],  (const float*)d_in[9],  (const float*)d_in[15]};
  const float* aw[3] = {(const float*)d_in[4],  (const float*)d_in[10], (const float*)d_in[16]};
  const float* ab[3] = {(const float*)d_in[5],  (const float*)d_in[11], (const float*)d_in[17]};
  const float* rw[3] = {(const float*)d_in[6],  (const float*)d_in[12], (const float*)d_in[18]};
  const float* rb[3] = {(const float*)d_in[7],  (const float*)d_in[13], (const float*)d_in[19]};
  const float* gg[2] = {(const float*)d_in[20], (const float*)d_in[22]};
  const float* bt[2] = {(const float*)d_in[21], (const float*)d_in[23]};

  char* ws = (char*)d_ws;
  size_t off = 0;
  auto carve = [&](size_t bytes) {
    char* p = ws + off;
    off += (bytes + 255) & ~(size_t)255;
    return p;
  };

  float*    DIS  = (float*)carve((size_t)N * 4);
  float*    XA   = (float*)carve((size_t)N * H_LD * 4);
  float*    XCAT = (float*)carve((size_t)N * XCAT_LD * 4);
  float*    HBUF = (float*)carve((size_t)N * H_LD * 4);
  float*    AGG  = (float*)carve((size_t)N * H_LD * 4);
  float*    TBUF = (float*)carve((size_t)N * T_LD * 4);          // t; reused as XB
  _Float16* XH   = (_Float16*)carve((size_t)N * XH_LD * 2);
  _Float16* TH   = (_Float16*)carve((size_t)(N + 16) * TH_LD * 2); // padded U|V|Z
  _Float16* WT   = (_Float16*)carve((size_t)448 * WT_LD * 2);
  float*    VTZF = (float*)carve(112 * 112 * 4);
  _Float16* VTZH = (_Float16*)carve(128 * 128 * 2);
  float*    CS   = (float*)carve(256 * 4);
  float*    NFI  = (float*)carve(256);
  float*    BNSC = (float*)carve(256 * 4);
  float*    BNSH = (float*)carve(256 * 4);
  float*    XB   = TBUF;  // alias: t fully consumed before readout GEMM writes

  auto gblocks = [](long total, int per) {
    long b = (total + per - 1) / per;
    if (b < 1) b = 1; if (b > 4096) b = 4096;
    return (int)b;
  };

  auto gemm = [&](const _Float16* A, int lda, const _Float16* B, int ldb,
                  float* C, int ldc, const float* bias, const float* scale,
                  int M, int Nstore, int strips, int Kd, int relu) {
    int waves  = (M >> 4) * strips;
    int blocks = (waves + 7) / 8;   // 8 waves / block
    k_gemm_fast<<<dim3(blocks), 256, 0, stream>>>(
        A, lda, B, ldb, C, ldc, bias, scale, M, Nstore, strips, Kd, relu);
  };

  // ---- degree normalization (in-degree + self loop) ----
  k_fill_f32<<<gblocks(N, 256), 256, 0, stream>>>(DIS, 1.0f, N);
  k_deg_edges<<<gblocks(E, 256), 256, 0, stream>>>(edst, DIS, E);
  k_rsqrt_ip<<<gblocks(N, 256), 256, 0, stream>>>(DIS, N);
  // zero the 16 padded tail rows of TH once (stay zero across layers)
  k_fill_f32<<<gblocks(16 * TH_LD / 2, 256), 256, 0, stream>>>(
      (float*)(TH + (size_t)N * TH_LD), 0.0f, 16 * TH_LD / 2);

  for (int L = 0; L < 3; ++L) {
    const int din = (L == 0) ? 128 : 256;
    const float* xin = (L == 0) ? x0 : XA;

    // x -> f16
    k_f32_to_f16_pad<<<4096, 256, 0, stream>>>(xin, din, din, XH, XH_LD, din, N);

    // h = x @ cw (bias + relu applied after aggregation)
    k_w_transpose_f16<<<gblocks((long)256 * WT_LD, 256), 256, 0, stream>>>(
        cw[L], din, 256, 256, WT, WT_LD);
    gemm(XH, XH_LD, WT, WT_LD, HBUF, H_LD, nullptr, nullptr, N, 256, 4, din, 0);

    // t = relu(x @ aw + ab)   (N=400 -> 25 tiles -> 7 strips, B padded to 448)
    k_w_transpose_f16<<<gblocks((long)448 * WT_LD, 256), 256, 0, stream>>>(
        aw[L], din, 400, 448, WT, WT_LD);
    gemm(XH, XH_LD, WT, WT_LD, TBUF, T_LD, ab[L], nullptr, N, 400, 7, din, 1);

    // GCN aggregation (bandwidth-bound)
    k_fill_f32<<<4096, 256, 0, stream>>>(AGG, 0.0f, (long)N * H_LD);
    k_scatter<<<8192, 256, 0, stream>>>(HBUF, DIS, esrc, edst, AGG, E);
    k_local_fin<<<4096, 256, 0, stream>>>(AGG, HBUF, DIS, cb[L], XCAT, N);

    // low-rank global path
    k_colsum<<<200, 256, 0, stream>>>(TBUF, CS, N);
    k_nf<<<1, 128, 0, stream>>>(CS, NFI, N);
    k_pack_t<<<4096, 256, 0, stream>>>(TBUF, TH, N);
    k_fill_f32<<<gblocks(112 * 112, 256), 256, 0, stream>>>(VTZF, 0.0f, 112 * 112);
    {
      // VtZ: 49 tiles x 63 K-chunks, Kd = 50016 (zero tail rows)
      int Kd = ((N + 31) / 32) * 32 + ((N % 32) ? 0 : 16);  // N=50000 -> 50016
      Kd = ((N + 16 + 31) / 32) * 32;                        // robust: 50016
      int kchunk = 800;
      int ky = (Kd + kchunk - 1) / kchunk;
      k_gemm_vtz<<<dim3(7, ky), 256, 0, stream>>>(TH, VTZF, kchunk, Kd);
    }
    k_conv_vtz<<<gblocks(128 * 128, 256), 256, 0, stream>>>(VTZF, VTZH);
    // xcat[:,0:100] = (U @ VtZ) * inv_nf   (K padded to 128, N padded to 128)
    gemm(TH, TH_LD, VTZH, 128, XCAT, XCAT_LD, nullptr, NFI, N, 100, 2, 128, 0);
    k_copy_T<<<gblocks((long)N * 100, 256), 256, 0, stream>>>(TBUF, XCAT, N);

    // readout: act(concat @ rw + rb), K padded 456 -> 480
    k_f32_to_f16_pad<<<4096, 256, 0, stream>>>(XCAT, XCAT_LD, 456, XH, XH_LD, 480, N);
    const int dw = (L == 2) ? 128 : 256;
    k_w_transpose_f16<<<gblocks((long)dw * WT_LD, 256), 256, 0, stream>>>(
        rw[L], 456, dw, dw, WT, WT_LD);
    float* outp = (L == 2) ? (float*)d_out : XB;
    gemm(XH, XH_LD, WT, WT_LD, outp, dw, rb[L], nullptr, N, dw, dw >> 6, 480,
         (L < 2) ? 1 : 0);

    if (L < 2) {
      k_bn_stats<<<256, 256, 0, stream>>>(XB, N, 256, gg[L], bt[L], BNSC, BNSH);
      k_bn_apply<<<4096, 256, 0, stream>>>(XB, BNSC, BNSH, XA, (long)N * 256, 256);
    }
  }
}
